// MultiHeadedAttention_59966333386766
// MI455X (gfx1250) — compile-verified, matrix operand order checked by reading, not run on hardware
//
#include <hip/hip_runtime.h>

typedef __attribute__((ext_vector_type(16))) __bf16 v16bf;
typedef __attribute__((ext_vector_type(8)))  __bf16 v8bf;
typedef __attribute__((ext_vector_type(8)))  float  v8f;

#define LOG2E 1.44269504f

// ---- dims (match setup_inputs) ----
#define Bn   2
#define Sn   2048
#define En   1024
#define Hn   16
#define Dn   64
#define Mn   (Bn * Sn)      // 4096
#define NQKV (3 * Hn * Dn)  // 3072

__device__ inline v8f wmma_bf16(v16bf a, v16bf b, v8f c) {
  return __builtin_amdgcn_wmma_f32_16x16x32_bf16(
      /*neg_a=*/false, a, /*neg_b=*/false, b,
      /*c_mod=*/(short)0, c, /*reuse_a=*/false, /*reuse_b=*/false);
}

// A-fragment layout for 16-bit 16x32 (also used for B via transposed operand):
// lane L<16 : row row0+L,   K chunks {k0..k0+7, k0+16..k0+23}
// lane L>=16: row row0+L-16,K chunks {k0+8..k0+15, k0+24..k0+31}
__device__ inline v16bf load_frag(const __bf16* base, int row0, int ld, int k0) {
  int lane = threadIdx.x & 31;
  int r = lane & 15, hf = lane >> 4;
  const __bf16* p = base + (long)(row0 + r) * ld + k0 + hf * 8;
  v8bf lo = *(const v8bf*)p;
  v8bf hi = *(const v8bf*)(p + 16);
  return __builtin_shufflevector(lo, hi, 0,1,2,3,4,5,6,7,8,9,10,11,12,13,14,15);
}

// ---- VALU-pipe 16-lane butterfly reductions via v_permlane16_b32 ----
__device__ inline float pl16(float x, unsigned s0, unsigned s1) {
  unsigned u = __float_as_uint(x);
  unsigned r = __builtin_amdgcn_permlane16(u, u, s0, s1, false, false);
  return __uint_as_float(r);
}
__device__ inline float rowmax16(float x) {
  x = fmaxf(x, pl16(x, 0x67452301u, 0xEFCDAB89u));   // xor 1
  x = fmaxf(x, pl16(x, 0x54761032u, 0xDCFE98BAu));   // xor 2
  x = fmaxf(x, pl16(x, 0x32107654u, 0xBA98FEDCu));   // xor 4
  x = fmaxf(x, pl16(x, 0xFEDCBA98u, 0x76543210u));   // xor 8
  return x;
}
__device__ inline float rowsum16(float x) {
  x += pl16(x, 0x67452301u, 0xEFCDAB89u);
  x += pl16(x, 0x54761032u, 0xDCFE98BAu);
  x += pl16(x, 0x32107654u, 0xBA98FEDCu);
  x += pl16(x, 0xFEDCBA98u, 0x76543210u);
  return x;
}

// ---- async global->LDS tile loader (ASYNCcnt path) ----
// Loads a 64x32 bf16 tile (row-major, ld=32 in LDS) with 16B chunks.
// 128 threads x 2 chunks each = 256 chunks.
__device__ inline void async_tile_load(const __bf16* g, long row0, int ldg, int k0,
                                       __bf16* lds_base) {
  int tid = threadIdx.x;
#pragma unroll
  for (int c = tid; c < 256; c += 128) {
    int row = c >> 2;
    int cp  = (c & 3) * 8;
    unsigned lo = (unsigned)(size_t)(lds_base + row * 32 + cp);  // low 32 bits = LDS offset
    const __bf16* gp = g + (row0 + row) * (long)ldg + k0 + cp;
    asm volatile("global_load_async_to_lds_b128 %0, %1, off"
                 :: "v"(lo), "v"(gp) : "memory");
  }
}
__device__ inline void wait_async0() {
  asm volatile("s_wait_asynccnt 0x0" ::: "memory");
}

// -------------------- fp32 -> bf16 conversion --------------------
__global__ void cvt_bf16_kernel(const float* __restrict__ in, __bf16* __restrict__ out, int n) {
  int i = blockIdx.x * blockDim.x + threadIdx.x;
  if (i < n) out[i] = (__bf16)in[i];
}

// -------------------- GEMM: C = A(MxK) * B^T (B is NxK row-major) --------------------
// LDS double-buffered tiles staged with global_load_async_to_lds_b128.
// MODE 0: N=3072, scatter into q[B,H,S,D], k[B,H,S,D], vT[B,H,D,S] (bf16)
// MODE 1: N=1024, plain fp32 store to out[M,N]
template <int MODE>
__global__ __launch_bounds__(128) void gemm_bf16_kernel(
    const __bf16* __restrict__ A, const __bf16* __restrict__ Bm, int K,
    __bf16* __restrict__ q_out, __bf16* __restrict__ k_out, __bf16* __restrict__ vT_out,
    float* __restrict__ f_out) {
  __shared__ __align__(16) __bf16 tA[2][64 * 32];
  __shared__ __align__(16) __bf16 tB[2][64 * 32];

  int lane = threadIdx.x & 31;
  int wave = threadIdx.x >> 5;
  long m_blk = blockIdx.y * 64;
  long n_blk = blockIdx.x * 64;
  int wm = (wave >> 1) * 32;     // wave's row offset within block tile
  int wn = (wave & 1) * 32;      // wave's col offset within block tile

  v8f z = {0.f,0.f,0.f,0.f,0.f,0.f,0.f,0.f};
  v8f acc[2][2];
  acc[0][0] = z; acc[0][1] = z; acc[1][0] = z; acc[1][1] = z;

  // prologue: stage first k-tile
  async_tile_load(A,  m_blk, K, 0, tA[0]);
  async_tile_load(Bm, n_blk, K, 0, tB[0]);
  wait_async0();
  __syncthreads();

  int buf = 0;
  for (int k0 = 0; k0 < K; k0 += 32) {
    if (k0 + 32 < K) {   // prefetch next tile into the other buffer
      async_tile_load(A,  m_blk, K, k0 + 32, tA[buf ^ 1]);
      async_tile_load(Bm, n_blk, K, k0 + 32, tB[buf ^ 1]);
    }
    v16bf a0 = load_frag(tA[buf], wm,      32, 0);
    v16bf a1 = load_frag(tA[buf], wm + 16, 32, 0);
    v16bf b0 = load_frag(tB[buf], wn,      32, 0);
    v16bf b1 = load_frag(tB[buf], wn + 16, 32, 0);
    acc[0][0] = wmma_bf16(a0, b0, acc[0][0]);
    acc[0][1] = wmma_bf16(a0, b1, acc[0][1]);
    acc[1][0] = wmma_bf16(a1, b0, acc[1][0]);
    acc[1][1] = wmma_bf16(a1, b1, acc[1][1]);

    wait_async0();
    __syncthreads();
    buf ^= 1;
  }

  int r15 = lane & 15, hf = lane >> 4;
#pragma unroll
  for (int sm = 0; sm < 2; ++sm)
#pragma unroll
    for (int sn = 0; sn < 2; ++sn)
#pragma unroll
      for (int r = 0; r < 8; ++r) {
        int m = (int)m_blk + wm + sm * 16 + r + hf * 8;   // token row
        int n = (int)n_blk + wn + sn * 16 + r15;          // feature col
        float c = acc[sm][sn][r];
        if (MODE == 0) {
          int which = n >> 10;                 // 0=q 1=k 2=v
          int f2 = n & 1023;
          int h = f2 >> 6, d = f2 & 63;
          int b = m >> 11, s = m & 2047;
          long bh = (long)(b * Hn + h);
          if (which == 0)      q_out[(bh * Sn + s) * Dn + d] = (__bf16)c;
          else if (which == 1) k_out[(bh * Sn + s) * Dn + d] = (__bf16)c;
          else                 vT_out[(bh * Dn + d) * Sn + s] = (__bf16)c;
        } else {
          f_out[(long)m * En + n] = c;
        }
      }
}

// -------------------- RoPE (in place, bf16 [B,H,S,D]); scale folded in --------------------
__global__ void rope_kernel(__bf16* __restrict__ t, float scale) {
  int i = blockIdx.x * blockDim.x + threadIdx.x;   // B*H*S*32 threads
  int d2 = i & 31;
  int s  = (i >> 5) & (Sn - 1);
  long base = (long)(i >> 5) * Dn;
  // inv_freq = 10000^{-2*d2/64} = exp2(-(d2/32) * log2(10000))
  float fr = exp2f(-(float)d2 * (13.2877123795f / 32.0f));
  float ang = (float)s * fr;
  float sn, cs;
  __sincosf(ang, &sn, &cs);
  float x1 = (float)t[base + 2 * d2];
  float x2 = (float)t[base + 2 * d2 + 1];
  t[base + 2 * d2]     = (__bf16)((x1 * cs - x2 * sn) * scale);
  t[base + 2 * d2 + 1] = (__bf16)((x2 * cs + x1 * sn) * scale);
}

// -------------------- Flash attention: one wave per 16-row query tile --------------------
__global__ __launch_bounds__(128) void flash_kernel(
    const __bf16* __restrict__ qb, const __bf16* __restrict__ kb,
    const __bf16* __restrict__ vT, __bf16* __restrict__ attnb) {
  __shared__ __align__(16) __bf16 plds[4][16 * 32];

  int wv = threadIdx.x >> 5;
  int w = blockIdx.x * 4 + wv;          // 0 .. B*H*(S/16)-1 = 4095
  int qt = w & 127;                     // query tile within (b,h)
  int bh = w >> 7;
  int q0 = qt * 16;

  const __bf16* qp = qb + (long)bh * Sn * Dn;
  const __bf16* kp = kb + (long)bh * Sn * Dn;
  const __bf16* vp = vT + (long)bh * Dn * Sn;

  int lane = threadIdx.x & 31;
  int r15 = lane & 15, hf = lane >> 4;

  v16bf qa0 = load_frag(qp, q0, Dn, 0);
  v16bf qa1 = load_frag(qp, q0, Dn, 32);

  v8f z = {0.f,0.f,0.f,0.f,0.f,0.f,0.f,0.f};
  v8f o[4];  o[0] = z; o[1] = z; o[2] = z; o[3] = z;
  float m_i[8], l_i[8];
#pragma unroll
  for (int r = 0; r < 8; ++r) { m_i[r] = -1e30f; l_i[r] = 0.f; }

  for (int j = 0; j <= q0 + 15; j += 32) {
    if (j + 32 <= q0 + 15) {
      __builtin_prefetch((const void*)(kp + (long)(j + 32) * Dn), 0, 0);
    }
    // S = Q @ K^T over 32 keys (two 16-key column groups)
    v16bf kA0 = load_frag(kp, j,      Dn, 0);
    v16bf kA1 = load_frag(kp, j,      Dn, 32);
    v16bf kB0 = load_frag(kp, j + 16, Dn, 0);
    v16bf kB1 = load_frag(kp, j + 16, Dn, 32);
    v8f s0 = z, s1 = z;
    s0 = wmma_bf16(qa0, kA0, s0); s0 = wmma_bf16(qa1, kA1, s0);
    s1 = wmma_bf16(qa0, kB0, s1); s1 = wmma_bf16(qa1, kB1, s1);

    float p0[8], p1[8];
#pragma unroll
    for (int r = 0; r < 8; ++r) {
      int mg = q0 + r + hf * 8;
      int ng0 = j + r15, ng1 = j + 16 + r15;
      float v0 = (ng0 <= mg) ? s0[r] : -1e30f;
      float v1 = (ng1 <= mg) ? s1[r] : -1e30f;
      float mx = rowmax16(fmaxf(v0, v1));
      float mnew = fmaxf(m_i[r], mx);
      float alpha = exp2f((m_i[r] - mnew) * LOG2E);
      float e0 = exp2f((v0 - mnew) * LOG2E);
      float e1 = exp2f((v1 - mnew) * LOG2E);
      float rs = rowsum16(e0 + e1);
      l_i[r] = l_i[r] * alpha + rs;
      m_i[r] = mnew;
      p0[r] = e0; p1[r] = e1;
#pragma unroll
      for (int c = 0; c < 4; ++c) o[c][r] *= alpha;
    }

    // transpose P (C-layout: lane=key col) -> A-layout via per-wave LDS bounce
#pragma unroll
    for (int r = 0; r < 8; ++r) {
      int m = r + hf * 8;
      plds[wv][m * 32 + r15]      = (__bf16)p0[r];
      plds[wv][m * 32 + 16 + r15] = (__bf16)p1[r];
    }
    v16bf pa;
    {
      const __bf16* lp = &plds[wv][r15 * 32 + hf * 8];
      v8bf lo = *(const v8bf*)lp;
      v8bf hi = *(const v8bf*)(lp + 16);
      pa = __builtin_shufflevector(lo, hi, 0,1,2,3,4,5,6,7,8,9,10,11,12,13,14,15);
    }

    // O += P @ V   (B frags are rows of vT: contiguous, ld = S)
#pragma unroll
    for (int c = 0; c < 4; ++c) {
      v16bf vb = load_frag(vp, c * 16, Sn, j);
      o[c] = wmma_bf16(pa, vb, o[c]);
    }
  }

  // normalize and write attn[B,S,H*D] bf16
  int b = bh >> 4, h = bh & 15;
#pragma unroll
  for (int r = 0; r < 8; ++r) {
    float inv = 1.0f / l_i[r];
    int m = r + hf * 8;
    long row = (long)(b * Sn + q0 + m);
#pragma unroll
    for (int c = 0; c < 4; ++c)
      attnb[row * (Hn * Dn) + h * Dn + c * 16 + r15] = (__bf16)(o[c][r] * inv);
  }
}

// -------------------- host side --------------------
extern "C" void kernel_launch(void* const* d_in, const int* in_sizes, int n_in,
                              void* d_out, int out_size, void* d_ws, size_t ws_size,
                              hipStream_t stream) {
  const float* x     = (const float*)d_in[0];   // [B,S,E]
  const float* w_qkv = (const float*)d_in[1];   // [3*H*D, E]
  const float* w_out = (const float*)d_in[2];   // [E, H*D]
  float* out = (float*)d_out;                   // [B,S,E]

  char* ws = (char*)d_ws;
  size_t off = 0;
  __bf16* xb    = (__bf16*)(ws + off); off += (size_t)Mn * En * 2;          // 8 MB
  __bf16* wqkvb = (__bf16*)(ws + off); off += (size_t)NQKV * En * 2;        // 6 MB
  __bf16* woutb = (__bf16*)(ws + off); off += (size_t)En * En * 2;          // 2 MB
  __bf16* qb    = (__bf16*)(ws + off); off += (size_t)Bn * Hn * Sn * Dn * 2;
  __bf16* kb    = (__bf16*)(ws + off); off += (size_t)Bn * Hn * Sn * Dn * 2;
  __bf16* vTb   = (__bf16*)(ws + off); off += (size_t)Bn * Hn * Dn * Sn * 2;
  __bf16* attnb = (__bf16*)(ws + off); off += (size_t)Mn * En * 2;
  (void)ws_size; (void)in_sizes; (void)n_in; (void)out_size;

  // 1) convert inputs to bf16
  {
    int n1 = Mn * En, n2 = NQKV * En, n3 = En * En;
    cvt_bf16_kernel<<<(n1 + 255) / 256, 256, 0, stream>>>(x, xb, n1);
    cvt_bf16_kernel<<<(n2 + 255) / 256, 256, 0, stream>>>(w_qkv, wqkvb, n2);
    cvt_bf16_kernel<<<(n3 + 255) / 256, 256, 0, stream>>>(w_out, woutb, n3);
  }

  // 2) QKV projection: qkv = x @ w_qkv^T, scattered into q/k/vT
  {
    dim3 grid(NQKV / 64, Mn / 64);
    gemm_bf16_kernel<0><<<grid, 128, 0, stream>>>(xb, wqkvb, En, qb, kb, vTb, nullptr);
  }

  // 3) RoPE on q (with 1/sqrt(D) folded in) and k
  {
    int n = Bn * Hn * Sn * 32;
    rope_kernel<<<n / 256, 256, 0, stream>>>(qb, 0.125f);
    rope_kernel<<<n / 256, 256, 0, stream>>>(kb, 1.0f);
  }

  // 4) causal flash attention
  {
    int tiles = Bn * Hn * (Sn / 16);   // 4096 wave-tiles, 4 per block
    flash_kernel<<<tiles / 4, 128, 0, stream>>>(qb, kb, vTb, attnb);
  }

  // 5) output projection: out = attn @ w_out^T  (fp32 result)
  {
    dim3 grid(En / 64, Mn / 64);
    gemm_bf16_kernel<1><<<grid, 128, 0, stream>>>(attnb, woutb, En, nullptr, nullptr, nullptr, out);
  }
}
